// ReformQwen3OmniMoeVisionAttention_22840636081003
// MI455X (gfx1250) — matrix operand order repaired and checked by reading, not compiled, not run on hardware
//
#include <hip/hip_runtime.h>
#include <hip/hip_bf16.h>

#define S_LEN 3072
#define DIM   1280
#define NH    16
#define HD    80
#define HDH   40      // HD/2
#define HDP   96      // HD padded to multiple of 32 for bf16 WMMA K-steps
#define NSEG  6
#define SEGLEN 512
#define QB    32      // query rows per attention block (scores = 32x512 f32 = 64KB LDS)
#define BK    64      // GEMM K-step (2 WMMA k-substeps per staged tile)

typedef __bf16 bhalf;
typedef __attribute__((ext_vector_type(8)))  __bf16 v8bf;
typedef __attribute__((ext_vector_type(16))) __bf16 v16bf;
typedef __attribute__((ext_vector_type(8)))  float  v8f;
typedef __attribute__((ext_vector_type(4)))  unsigned int v4u;
typedef __attribute__((ext_vector_type(8)))  int v8i;
typedef __attribute__((ext_vector_type(4)))  int v4i;

__device__ __forceinline__ v16bf cat16(v8bf lo, v8bf hi) {
  v16bf r;
#pragma unroll
  for (int j = 0; j < 8; ++j) { r[j] = lo[j]; r[8 + j] = hi[j]; }
  return r;
}

__device__ __forceinline__ v8f wmma_bf16f32(v16bf a, v16bf b, v8f c) {
  // D = A(16x32 bf16) x B(32x16 bf16) + C(16x16 f32)
  return __builtin_amdgcn_wmma_f32_16x16x32_bf16(
      false, a, false, b, (short)0, c, false, false);
}

// Low 32 bits of a flat LDS pointer are the LDS byte offset (ISA: flat->LDS
// mapping truncates to addr[31:0]).
__device__ __forceinline__ unsigned lds_addr_of(const void* p) {
  return (unsigned)(unsigned long long)(uintptr_t)p;
}

// TDM 2D tile load: tile_rows x tile_k bf16 elements from a row-major matrix
// with row stride `row_stride` elems into LDS (packed row-major, stride tile_k).
// Issued by whichever wave executes it; completion tracked with TENSORcnt.
__device__ __forceinline__ void tdm_load_2d(bhalf* lds, const bhalf* gsrc,
                                            unsigned tile_k, unsigned tile_rows,
                                            unsigned tensor_k, unsigned tensor_rows,
                                            unsigned row_stride) {
#if __has_builtin(__builtin_amdgcn_tensor_load_to_lds)
  unsigned long long ga = (unsigned long long)(uintptr_t)gsrc;
  v4u g0;
  g0[0] = 1u;                                        // count=1, user descriptor
  g0[1] = lds_addr_of(lds);                          // lds_addr (bytes)
  g0[2] = (unsigned)(ga & 0xffffffffu);              // global_addr[31:0]
  g0[3] = (unsigned)((ga >> 32) & 0x01ffffffu)       // global_addr[56:32]
          | (2u << 30);                              // type=2 ("image")
  v8i g1;
  g1[0] = (int)(1u << 16);                           // wg_mask=0 | data_size=1 (2B)
  g1[1] = (int)((tensor_k & 0xffffu) << 16);         // tensor_dim0[15:0]
  g1[2] = (int)(((tensor_k >> 16) & 0xffffu) | ((tensor_rows & 0xffffu) << 16));
  g1[3] = (int)(((tensor_rows >> 16) & 0xffffu) | ((tile_k & 0xffffu) << 16));
  g1[4] = (int)(tile_rows & 0xffffu);                // tile_dim1 | tile_dim2=0
  g1[5] = (int)row_stride;                           // tensor_dim0_stride[31:0]
  g1[6] = 0;                                         // stride0 hi | stride1 lo
  g1[7] = 0;
  v4i z4 = {0, 0, 0, 0};
  v8i z8 = {0, 0, 0, 0, 0, 0, 0, 0};
  // this toolchain declares the 6-arg form: (g0, g1, g2, g3, extra, cpol)
  __builtin_amdgcn_tensor_load_to_lds(g0, g1, z4, z4, z8, 0);
#else
  // fallback: cooperative copy by the issuing wave's 32 lanes
  int lane = threadIdx.x & 31;
  for (unsigned r = 0; r < tile_rows; ++r)
    for (unsigned c = lane * 8u; c < tile_k; c += 32u * 8u)
      *(v8bf*)&lds[r * tile_k + c] = *(const v8bf*)&gsrc[r * row_stride + c];
#endif
}

__device__ __forceinline__ void tdm_wait0() {
#if __has_builtin(__builtin_amdgcn_s_wait_tensorcnt)
  __builtin_amdgcn_s_wait_tensorcnt(0);
#endif
}

// ---------------- conversion kernel ----------------

__global__ __launch_bounds__(256) void cvt_f32_bf16(const float* __restrict__ x,
                                                    bhalf* __restrict__ y, int n) {
  int i = blockIdx.x * 256 + threadIdx.x;
  if (i < n) y[i] = (bhalf)x[i];
}

// ---------------- tiled bf16 WMMA GEMM: C = A(MxK) * B(NxK)^T + bias ----------------
// A: M x K row-major activations; B: N x K row-major weights (native layout!).
// block = 256 threads = 8 waves, tile 128x64, per-wave 32x32 (2x2 WMMA tiles).
// Double-buffered LDS staging via Tensor Data Mover, wave 0 drives the DMA.

__global__ __launch_bounds__(256) void gemm_bias_bf16(
    const bhalf* __restrict__ A, const bhalf* __restrict__ B,
    const float* __restrict__ bias, float* __restrict__ C,
    int M, int N, int K) {
  __shared__ __align__(16) bhalf lA[2][128 * BK];   // 2 x 16KB
  __shared__ __align__(16) bhalf lB[2][64 * BK];    // 2 x 8KB
  const int tid   = threadIdx.x;
  const int wave  = tid >> 5;
  const int lane  = tid & 31;
  const int lhalf = lane >> 4;     // 0: lanes 0-15, 1: lanes 16-31
  const int r16   = lane & 15;
  const int mwave = wave & 3;      // 4 wave-rows
  const int nwave = wave >> 2;     // 2 wave-cols
  const int bm0 = blockIdx.x * 128;
  const int bn0 = blockIdx.y * 64;
  const int nk = K / BK;

  v8f acc[2][2] = {};

  if (wave == 0) {
    tdm_load_2d(&lA[0][0], A + bm0 * K, BK, 128, K, M, K);
    tdm_load_2d(&lB[0][0], B + bn0 * K, BK, 64, K, N, K);
  }

  for (int ks = 0; ks < nk; ++ks) {
    const int cur = ks & 1;
    if (wave == 0) tdm_wait0();          // staged tiles for this step landed
    __syncthreads();                     // publish to all waves
    if (wave == 0 && ks + 1 < nk) {      // prefetch next step into other buffer
      int k0 = (ks + 1) * BK;
      tdm_load_2d(&lA[cur ^ 1][0], A + bm0 * K + k0, BK, 128, K, M, K);
      tdm_load_2d(&lB[cur ^ 1][0], B + bn0 * K + k0, BK, 64, K, N, K);
    }
#pragma unroll
    for (int kk = 0; kk < 2; ++kk) {     // two 32-deep WMMA substeps
      v16bf af[2], bfrag[2];
#pragma unroll
      for (int mi = 0; mi < 2; ++mi) {
        // A 16x32 frag: lanes 0-15 row=r16 K 0-7/16-23; lanes 16-31 K 8-15/24-31
        const bhalf* p = &lA[cur][(mwave * 32 + mi * 16 + r16) * BK + kk * 32 + lhalf * 8];
        af[mi] = cat16(*(const v8bf*)p, *(const v8bf*)(p + 16));
      }
#pragma unroll
      for (int ni = 0; ni < 2; ++ni) {
        // B 32x16 frag: lanes 0-15 col=r16 K 0-7/16-23; lanes 16-31 K 8-15/24-31
        const bhalf* p = &lB[cur][(nwave * 32 + ni * 16 + r16) * BK + kk * 32 + lhalf * 8];
        bfrag[ni] = cat16(*(const v8bf*)p, *(const v8bf*)(p + 16));
      }
#pragma unroll
      for (int mi = 0; mi < 2; ++mi)
#pragma unroll
        for (int ni = 0; ni < 2; ++ni)
          acc[mi][ni] = wmma_bf16f32(af[mi], bfrag[ni], acc[mi][ni]);
    }
    __syncthreads();                     // all reads done before buffer reuse
  }

  // epilogue: C/D layout — lane -> col (r16), VGPR g -> row g + 8*lhalf
#pragma unroll
  for (int mi = 0; mi < 2; ++mi)
#pragma unroll
    for (int ni = 0; ni < 2; ++ni) {
      int col = bn0 + nwave * 32 + ni * 16 + r16;
      float bv = bias[col];
      int base = (bm0 + mwave * 32 + mi * 16 + 8 * lhalf) * N + col;
#pragma unroll
      for (int g = 0; g < 8; ++g)
        C[base + g * N] = acc[mi][ni][g] + bv;
    }
}

// ---------------- RoPE + repack to head-major padded bf16 ----------------
// Qb,Kb: (H, S, HDP) row-major;  Vt: (H, HDP, S) transposed

__global__ __launch_bounds__(256) void rope_pack(
    const float* __restrict__ Qf, const float* __restrict__ Kf,
    const float* __restrict__ Vf, const float* __restrict__ cosb,
    const float* __restrict__ sinb, bhalf* __restrict__ Qb,
    bhalf* __restrict__ Kb, bhalf* __restrict__ Vt) {
  int idx = blockIdx.x * 256 + threadIdx.x;
  const int total = NH * S_LEN * HDP;
  if (idx >= total) return;
  int d = idx % HDP;
  int t = (idx / HDP) % S_LEN;
  int h = idx / (HDP * S_LEN);
  float qv = 0.f, kv = 0.f, vv = 0.f;
  if (d < HD) {
    float c = cosb[t * HD + d];
    float s = sinb[t * HD + d];
    int base = t * DIM + h * HD;
    float q = Qf[base + d];
    float k = Kf[base + d];
    float q2, k2;
    if (d < HDH) { q2 = -Qf[base + d + HDH]; k2 = -Kf[base + d + HDH]; }
    else         { q2 =  Qf[base + d - HDH]; k2 =  Kf[base + d - HDH]; }
    qv = q * c + q2 * s;
    kv = k * c + k2 * s;
    vv = Vf[base + d];
  }
  Qb[(h * S_LEN + t) * HDP + d] = (bhalf)qv;
  Kb[(h * S_LEN + t) * HDP + d] = (bhalf)kv;
  Vt[(h * HDP + d) * S_LEN + t] = (bhalf)vv;
}

// ---------------- block-diagonal attention ----------------
// grid = (SEGLEN/QB, NH, NSEG); block = 256 threads (8 waves)
// dynamic LDS: scores 32x512 f32 = 64KB exactly

__global__ __launch_bounds__(256) void attn_kernel(
    const bhalf* __restrict__ Qb, const bhalf* __restrict__ Kb,
    const bhalf* __restrict__ Vt, bhalf* __restrict__ Ob) {
  extern __shared__ float sc[];    // [QB][SEGLEN]
  const float SCALE = 0.111803398874989484f;  // 80^-0.5
  const int tid   = threadIdx.x;
  const int wave  = tid >> 5;
  const int lane  = tid & 31;
  const int lhalf = lane >> 4;
  const int r16   = lane & 15;
  const int qblk = blockIdx.x;
  const int h    = blockIdx.y;
  const int seg  = blockIdx.z;
  const int s0 = seg * SEGLEN + qblk * QB;

  const bhalf* Qh = Qb + h * S_LEN * HDP;
  const bhalf* Kh = Kb + h * S_LEN * HDP;

  // ---- phase 1: scores = Q(32x96) * K^T -> 32x512, scaled ----
  const int mtile = wave & 1;            // 2 m-tiles of 16
  const int nbase = (wave >> 1) * 8;     // each wave covers 8 of 32 n-tiles
  v16bf afr[3];
  {
    int qrow = s0 + mtile * 16 + r16;
#pragma unroll
    for (int kk = 0; kk < 3; ++kk) {
      const bhalf* p = Qh + qrow * HDP + kk * 32 + lhalf * 8;
      afr[kk] = cat16(*(const v8bf*)p, *(const v8bf*)(p + 16));
    }
  }
  for (int j = 0; j < 8; ++j) {
    int ntile = nbase + j;
    int tcol = seg * SEGLEN + ntile * 16 + r16;   // key token = B column
    if (j + 1 < 8)                                 // prefetch next key row
      __builtin_prefetch(Kh + (tcol + 16) * HDP, 0, 0);
    v8f acc = {};
#pragma unroll
    for (int kk = 0; kk < 3; ++kk) {
      const bhalf* p = Kh + tcol * HDP + kk * 32 + lhalf * 8;
      v16bf bfrag = cat16(*(const v8bf*)p, *(const v8bf*)(p + 16));
      acc = wmma_bf16f32(afr[kk], bfrag, acc);
    }
#pragma unroll
    for (int g = 0; g < 8; ++g)
      sc[(mtile * 16 + g + 8 * lhalf) * SEGLEN + ntile * 16 + r16] = acc[g] * SCALE;
  }
  __syncthreads();

  // ---- phase 2: row softmax via wave32 shuffles (4 rows per wave) ----
#pragma unroll
  for (int i = 0; i < 4; ++i) {
    int r = wave * 4 + i;
    float* row = sc + r * SEGLEN;
    float m = -3.4e38f;
    for (int c = lane; c < SEGLEN; c += 32) m = fmaxf(m, row[c]);
#pragma unroll
    for (int off = 16; off > 0; off >>= 1) m = fmaxf(m, __shfl_xor(m, off, 32));
    float sum = 0.f;
    for (int c = lane; c < SEGLEN; c += 32) {
      float e = __expf(row[c] - m);
      row[c] = e;
      sum += e;
    }
#pragma unroll
    for (int off = 16; off > 0; off >>= 1) sum += __shfl_xor(sum, off, 32);
    float inv = 1.f / sum;
    for (int c = lane; c < SEGLEN; c += 32) row[c] *= inv;
  }
  __syncthreads();

  // ---- phase 3: O = P(32x512) * V(512x96), keep d<80 ----
  for (int t = wave; t < 12; t += 8) {           // 2 m-tiles x 6 n-tiles
    int mt = t / 6, nt = t % 6;
    int dcol = nt * 16 + r16;
    const bhalf* Vcol = Vt + (h * HDP + dcol) * S_LEN + seg * SEGLEN;
    v8f acc = {};
    for (int k0 = 0; k0 < SEGLEN; k0 += 32) {
      const float* ap = &sc[(mt * 16 + r16) * SEGLEN + k0 + lhalf * 8];
      v16bf a;
#pragma unroll
      for (int j = 0; j < 8; ++j) { a[j] = (bhalf)ap[j]; a[8 + j] = (bhalf)ap[16 + j]; }
      const bhalf* bp = Vcol + k0 + lhalf * 8;
      v16bf bfrag = cat16(*(const v8bf*)bp, *(const v8bf*)(bp + 16));
      acc = wmma_bf16f32(a, bfrag, acc);
    }
    if (dcol < HD) {
#pragma unroll
      for (int g = 0; g < 8; ++g) {
        int srow = s0 + mt * 16 + g + 8 * lhalf;
        Ob[srow * DIM + h * HD + dcol] = (bhalf)acc[g];
      }
    }
  }
}

// ---------------- host launch ----------------

extern "C" void kernel_launch(void* const* d_in, const int* in_sizes, int n_in,
                              void* d_out, int out_size, void* d_ws, size_t ws_size,
                              hipStream_t stream) {
  (void)in_sizes; (void)n_in; (void)out_size; (void)ws_size;
  const float* hidden = (const float*)d_in[0];
  const float* cosb   = (const float*)d_in[1];
  const float* sinb   = (const float*)d_in[2];
  const float* wq     = (const float*)d_in[3];
  const float* bq     = (const float*)d_in[4];
  const float* wk     = (const float*)d_in[5];
  const float* bk     = (const float*)d_in[6];
  const float* wv     = (const float*)d_in[7];
  const float* bvp    = (const float*)d_in[8];
  const float* wo     = (const float*)d_in[9];
  const float* bo     = (const float*)d_in[10];
  // d_in[11] = cu_seqlens: uniform S/NSEG segments, handled structurally
  float* out = (float*)d_out;

  char* wsp = (char*)d_ws;
  auto take = [&](size_t bytes) {
    char* p = wsp;
    wsp += (bytes + 255) & ~(size_t)255;
    return (void*)p;
  };
  bhalf* Xb  = (bhalf*)take((size_t)S_LEN * DIM * 2);
  bhalf* Wqb = (bhalf*)take((size_t)DIM * DIM * 2);
  bhalf* Wkb = (bhalf*)take((size_t)DIM * DIM * 2);
  bhalf* Wvb = (bhalf*)take((size_t)DIM * DIM * 2);
  bhalf* Wob = (bhalf*)take((size_t)DIM * DIM * 2);
  float* Qf  = (float*)take((size_t)S_LEN * DIM * 4);
  float* Kf  = (float*)take((size_t)S_LEN * DIM * 4);
  float* Vf  = (float*)take((size_t)S_LEN * DIM * 4);
  bhalf* Qbb = (bhalf*)take((size_t)NH * S_LEN * HDP * 2);
  bhalf* Kbb = (bhalf*)take((size_t)NH * S_LEN * HDP * 2);
  bhalf* Vtb = (bhalf*)take((size_t)NH * HDP * S_LEN * 2);
  bhalf* Ob  = (bhalf*)take((size_t)S_LEN * DIM * 2);

  cvt_f32_bf16<<<(S_LEN * DIM + 255) / 256, 256, 0, stream>>>(hidden, Xb, S_LEN * DIM);
  cvt_f32_bf16<<<(DIM * DIM + 255) / 256, 256, 0, stream>>>(wq, Wqb, DIM * DIM);
  cvt_f32_bf16<<<(DIM * DIM + 255) / 256, 256, 0, stream>>>(wk, Wkb, DIM * DIM);
  cvt_f32_bf16<<<(DIM * DIM + 255) / 256, 256, 0, stream>>>(wv, Wvb, DIM * DIM);
  cvt_f32_bf16<<<(DIM * DIM + 255) / 256, 256, 0, stream>>>(wo, Wob, DIM * DIM);

  dim3 gg(S_LEN / 128, DIM / 64);
  gemm_bias_bf16<<<gg, 256, 0, stream>>>(Xb, Wqb, bq, Qf, S_LEN, DIM, DIM);
  gemm_bias_bf16<<<gg, 256, 0, stream>>>(Xb, Wkb, bk, Kf, S_LEN, DIM, DIM);
  gemm_bias_bf16<<<gg, 256, 0, stream>>>(Xb, Wvb, bvp, Vf, S_LEN, DIM, DIM);

  rope_pack<<<(NH * S_LEN * HDP + 255) / 256, 256, 0, stream>>>(
      Qf, Kf, Vf, cosb, sinb, Qbb, Kbb, Vtb);

  attn_kernel<<<dim3(SEGLEN / QB, NH, NSEG), 256, QB * SEGLEN * sizeof(float), stream>>>(
      Qbb, Kbb, Vtb, Ob);

  gemm_bias_bf16<<<gg, 256, 0, stream>>>(Ob, Wob, bo, out, S_LEN, DIM, DIM);
}
